// TEDGN_52304111731111
// MI455X (gfx1250) — matrix operand order repaired and checked by reading, not compiled, not run on hardware
//
#include <hip/hip_runtime.h>
#include <hip/hip_bf16.h>
#include <math.h>

// Problem constants (match reference)
#define BB   128
#define TT   128
#define VV   64
#define DD   32
#define QQ   5
#define PLMK 768
#define KCAT 325          // 5*65 folded gate K
#define KC   352          // padded to multiple of 32
#define HID  200
#define HIDP 208
#define NCLS 2
#define ALPHA_C 0.5f

typedef __attribute__((ext_vector_type(16))) __bf16 v16bf;
typedef __attribute__((ext_vector_type(8)))  __bf16 v8bf;
typedef __attribute__((ext_vector_type(8)))  float  v8f;

// ---------------------------------------------------------------------------
// WMMA helpers (bf16 16x16x32, f32 accum) — wave32, CDNA5 VGPR layouts
// ---------------------------------------------------------------------------
__device__ __forceinline__ v8f wmma_bf16(v16bf a, v16bf b, v8f c) {
  return __builtin_amdgcn_wmma_f32_16x16x32_bf16(false, a, false, b,
                                                 (short)0, c, false, false);
}

// A-matrix 16x32 bf16 fragment from LDS (row-major, leading dim ld, ld%8==0).
// Lane l<16: M=l, K in {0..7}∪{16..23}; lane>=16: +8 K offset.
// Two contiguous 8-elem (16B) chunks -> 2x ds_load_b128.
__device__ __forceinline__ v16bf ldsA(const __bf16* s, int ld, int mOff, int kOff) {
  const int lane = threadIdx.x & 31;
  const __bf16* p = s + (mOff + (lane & 15)) * ld + kOff + ((lane & 16) ? 8 : 0);
  const v8bf lo = *(const v8bf*)p;
  const v8bf hi = *(const v8bf*)(p + 16);
  return __builtin_shufflevector(lo, hi, 0,1,2,3,4,5,6,7,8,9,10,11,12,13,14,15);
}

// B-matrix 32x16 bf16 fragment from LDS stored TRANSPOSED (N-major, K
// contiguous, leading dim ldK, ldK%8==0).  Same K order as ldsA:
// lane<16: K in {0..7}u{16..23}; lane>=16: +8 K offset. N = nOff + (lane&15).
__device__ __forceinline__ v16bf ldsBT(const __bf16* st, int ldK, int kOff, int nOff) {
  const int lane = threadIdx.x & 31;
  const __bf16* p = st + (nOff + (lane & 15)) * ldK + kOff + ((lane & 16) ? 8 : 0);
  const v8bf lo = *(const v8bf*)p;
  const v8bf hi = *(const v8bf*)(p + 16);
  return __builtin_shufflevector(lo, hi, 0,1,2,3,4,5,6,7,8,9,10,11,12,13,14,15);
}

// C/D 16x16 f32 store: VGPR r -> M = mOff + r + (lane>=16 ? 8:0), N = nOff + (lane&15)
__device__ __forceinline__ void stC(float* s, int ld, int mOff, int nOff, v8f c) {
  const int lane = threadIdx.x & 31;
  const int n  = nOff + (lane & 15);
  const int mb = mOff + ((lane & 16) ? 8 : 0);
#pragma unroll
  for (int r = 0; r < 8; ++r) s[(mb + r) * ld + n] = c[r];
}

// ===========================================================================
// Kernel 1: precompute  var_vector, adj(softmax), folded gate weights
// (stored transposed [o][k]), biases, static embedding.
// ===========================================================================
struct __align__(32) PreSmem {
  __bf16 plm[VV][PLMK];     // 96 KB  (A operand, row-major)
  __bf16 wT[VV][PLMK];      // 96 KB  (B operand, transposed: [n][k])
  float  h1[VV][VV];        // 16 KB
  float  gvec[VV][8];
  float  vv[VV][QQ];
};

__device__ void gemm64x64_k768(PreSmem& S, const float* bias, int wave, int lane) {
  for (int tile = wave; tile < 16; tile += 8) {
    const int mo = (tile & 3) * 16;
    const int no = (tile >> 2) * 16;
    v8f acc = {};
#pragma unroll
    for (int kc = 0; kc < PLMK; kc += 32)
      acc = wmma_bf16(ldsA(&S.plm[0][0], PLMK, mo, kc),
                      ldsBT(&S.wT[0][0], PLMK, kc, no), acc);
    const int n  = no + (lane & 15);
    const int mb = mo + ((lane & 16) ? 8 : 0);
#pragma unroll
    for (int r = 0; r < 8; ++r)
      S.h1[mb + r][n] = fmaxf(acc[r] + bias[n], 0.0f);
  }
}

__global__ __launch_bounds__(256, 1)
void k_pre(const float* plm, const float* Ws1, const float* bs1,
           const float* Ws2, const float* bs2,
           const float* Wg1, const float* bg1,
           const float* Wg2, const float* bg2,
           const float* Wr, const float* Wu, const float* Wc,
           const float* br, const float* bu, const float* bc,
           const float* Pst, const float* W_se, const float* b_se,
           float* g_vv, float* g_adj,
           __bf16* g_wr, __bf16* g_wu, __bf16* g_wc,
           float* g_beffr, float* g_beffu, float* g_beffc,
           float* g_se) {
  extern __shared__ char smem_raw[];
  PreSmem& S = *(PreSmem*)smem_raw;
  const int tid  = threadIdx.x;
  const int wave = tid >> 5;
  const int lane = tid & 31;

  for (int i = tid; i < VV * PLMK; i += 256) ((__bf16*)S.plm)[i] = (__bf16)plm[i];
  for (int i = tid; i < PLMK * VV; i += 256) {        // transpose Ws1 into wT
    const int k = i >> 6, n = i & 63;                 // coalesced global read
    S.wT[n][k] = (__bf16)Ws1[i];
  }
  __syncthreads();

  gemm64x64_k768(S, bs1, wave, lane);       // H1 = relu(plm @ Ws1 + bs1)
  __syncthreads();

  // var_vector = H1 @ Ws2 + bs2   (64x5) ; reload wT <- Wg1^T
  for (int i = tid; i < VV * QQ; i += 256) {
    const int v = i / QQ, q = i % QQ;
    float s = bs2[q];
    for (int k = 0; k < VV; ++k) s += S.h1[v][k] * Ws2[k * QQ + q];
    S.vv[v][q] = s;
    g_vv[i] = s;
  }
  for (int i = tid; i < PLMK * VV; i += 256) {
    const int k = i >> 6, n = i & 63;
    S.wT[n][k] = (__bf16)Wg1[i];
  }
  __syncthreads();

  gemm64x64_k768(S, bg1, wave, lane);       // G1 = relu(plm @ Wg1 + bg1)
  __syncthreads();

  // gvec = G1 @ Wg2 + bg2 (64x8)
  for (int i = tid; i < VV * 8; i += 256) {
    const int v = i >> 3, e = i & 7;
    float s = bg2[e];
    for (int k = 0; k < VV; ++k) s += S.h1[v][k] * Wg2[k * 8 + e];
    S.gvec[v][e] = s;
  }
  __syncthreads();
  if (tid < VV) {
    float nrm = 0.f;
#pragma unroll
    for (int e = 0; e < 8; ++e) nrm += S.gvec[tid][e] * S.gvec[tid][e];
    nrm = fmaxf(sqrtf(nrm), 1e-12f);
#pragma unroll
    for (int e = 0; e < 8; ++e) S.gvec[tid][e] /= nrm;
  }
  __syncthreads();
  // adj = softmax(gvec @ gvec^T) row-wise
  if (tid < VV) {
    float mx = -1e30f;
    for (int j = 0; j < VV; ++j) {
      float s = 0.f;
#pragma unroll
      for (int e = 0; e < 8; ++e) s += S.gvec[tid][e] * S.gvec[j][e];
      S.h1[tid][j] = s;
      mx = fmaxf(mx, s);
    }
    float den = 0.f;
    for (int j = 0; j < VV; ++j) { float ex = expf(S.h1[tid][j] - mx); S.h1[tid][j] = ex; den += ex; }
    for (int j = 0; j < VV; ++j) g_adj[tid * VV + j] = S.h1[tid][j] / den;
  }

  // Folded gate weights, stored TRANSPOSED: g_w[o*KC + k] = W[k][o], pad k>=325
  for (int i = tid; i < DD * KC; i += 256) {
    const int o = i / KC, k = i - o * KC;
    const bool live = (k < KCAT);
    g_wr[i] = (__bf16)(live ? Wr[k * DD + o] : 0.0f);
    g_wu[i] = (__bf16)(live ? Wu[k * DD + o] : 0.0f);
    g_wc[i] = (__bf16)(live ? Wc[k * DD + o] : 0.0f);
  }
  // Effective biases: beff[v][o] = sum_q vv[v][q] * b[q][o]
  for (int i = tid; i < VV * DD; i += 256) {
    const int v = i >> 5, o = i & 31;
    float sr = 0.f, su = 0.f, sc = 0.f;
#pragma unroll
    for (int q = 0; q < QQ; ++q) {
      const float vq = S.vv[v][q];
      sr += vq * br[q * DD + o];
      su += vq * bu[q * DD + o];
      sc += vq * bc[q * DD + o];
    }
    g_beffr[i] = sr; g_beffu[i] = su; g_beffc[i] = sc;
  }
  // static_emb = P_static @ W_se + b_se (128x64)
  for (int i = tid; i < BB * VV; i += 256) {
    const int b = i >> 6, j = i & 63;
    float s = b_se[j];
#pragma unroll
    for (int k = 0; k < 16; ++k) s += Pst[b * 16 + k] * W_se[k * VV + j];
    g_se[i] = s;
  }
}

// ===========================================================================
// Kernel 2: recurrent scan. One workgroup per batch, 8 waves, all state in LDS.
// float section first (size % 32 == 0), then 32B-aligned bf16 section.
// ===========================================================================
struct __align__(32) MainSmem {
  float  adj[VV][VV];        // 16 KB
  float  rarw[VV][VV];       // 16 KB  rarity_W cached
  float  h[VV][DD];
  float  hout[VV][DD];
  float  obs[VV][DD];
  float  comb[VV][80];       // comb GEMM output (cols 0..64 live)
  float  gr[VV][DD];
  float  gu[VV][DD];
  float  gc[VV][DD];
  float  beffr[VV][DD];
  float  beffu[VV][DD];
  float  beffc[VV][DD];
  float  emb1s[VV][DD];
  float  vv[VV][QQ];
  float  rar[VV];
  float  vto[VV];
  float  msk[VV];
  float  dats[VV];
  float  tes[DD];
  float  wvs[DD];
  float  bvs[DD];
  float  wps[DD];
  float  bps[DD];
  __bf16 adjb[VV][VV];       // cur_adj (A operand, row-major)
  __bf16 xhT[80][VV];        // [x | rar | h | pad] TRANSPOSED (B operand)
  __bf16 combq[VV][KC];      // scaled gate input (A operand, row-major)
  __bf16 wrT[DD][KC];        // gate weights transposed (B operands)
  __bf16 wuT[DD][KC];
  __bf16 wcT[DD][KC];
};

__global__ __launch_bounds__(256, 1)
void k_main(const float* P, const float* Pavg, const int* Plen, const float* Ptime,
            const float* w_val, const float* b_val,
            const float* w_per, const float* b_per,
            const float* w_lin, const float* b_lin,
            const float* emb1, const float* rarW,
            const float* g_adj, const float* g_vv,
            const __bf16* g_wr, const __bf16* g_wu, const __bf16* g_wc,
            const float* g_beffr, const float* g_beffu, const float* g_beffc,
            float* g_outsum) {
  extern __shared__ char smem_raw[];
  MainSmem& S = *(MainSmem*)smem_raw;
  const int b    = blockIdx.x;
  const int tid  = threadIdx.x;
  const int wave = tid >> 5;
  const int lane = tid & 31;

  for (int i = tid; i < VV * VV; i += 256) {
    ((float*)S.adj)[i]  = g_adj[i];
    ((float*)S.rarw)[i] = rarW[i];
  }
  for (int i = tid; i < VV * QQ; i += 256) ((float*)S.vv)[i] = g_vv[i];
  for (int i = tid; i < DD * KC; i += 256) {
    ((__bf16*)S.wrT)[i] = g_wr[i];
    ((__bf16*)S.wuT)[i] = g_wu[i];
    ((__bf16*)S.wcT)[i] = g_wc[i];
  }
  for (int i = tid; i < VV * DD; i += 256) {
    ((float*)S.h)[i] = 0.f; ((float*)S.hout)[i] = 0.f;
    ((float*)S.beffr)[i] = g_beffr[i];
    ((float*)S.beffu)[i] = g_beffu[i];
    ((float*)S.beffc)[i] = g_beffc[i];
    ((float*)S.emb1s)[i] = emb1[i];
  }
  if (tid < DD) {
    S.wvs[tid] = w_val[tid];
    S.bvs[tid] = b_val[tid];
    S.wps[tid] = (tid < DD - 1) ? w_per[tid] : 0.f;
    S.bps[tid] = (tid < DD - 1) ? b_per[tid] : 0.f;
  }
  // zero combq pad columns (k >= 325) and xhT pad rows (65..79) ONCE
  for (int i = tid; i < VV * KC; i += 256) {
    if ((i % KC) >= KCAT) ((__bf16*)S.combq)[i] = (__bf16)0.0f;
  }
  for (int i = tid; i < 15 * VV; i += 256) (&S.xhT[65][0])[i] = (__bf16)0.0f;
  if (tid < VV) {                       // var_total_obs[b, v]
    float s = 0.f;
    for (int t = 0; t < TT; ++t) s += P[((size_t)b * TT + t) * (2 * VV) + VV + tid];
    S.vto[tid] = s;
  }
  const int len = Plen[b];
  const float wl = w_lin[0], bl = b_lin[0];
  __syncthreads();

  for (int t = 0; t < TT; ++t) {
    // prefetch next step's rows into cache (global_prefetch_b8)
    if (t + 1 < TT) {
      const float* np = &P[((size_t)b * TT + (t + 1)) * (2 * VV)];
      const float* na = &Pavg[((size_t)b * TT + (t + 1)) * VV];
      if (tid < 4) __builtin_prefetch(np + tid * 32, 0, 0);
      if (tid == 4) __builtin_prefetch(na, 0, 0);
      if (tid == 5) __builtin_prefetch(na + 32, 0, 0);
    }
    const float tval = Ptime[b * TT + t];
    if (tid < VV) {
      S.dats[tid] = P[((size_t)b * TT + t) * (2 * VV) + tid];
      const float m = P[((size_t)b * TT + t) * (2 * VV) + VV + tid];
      S.msk[tid] = m;
      const float avg = Pavg[((size_t)b * TT + t) * VV + tid];
      S.rar[tid] = ALPHA_C * tanhf(avg / (S.vto[tid] + 1.0f));
    }
    if (tid >= 224) {                   // last wave: time_emb[d], v-independent
      const int d = tid - 224;
      S.tes[d] = (d == 0) ? (tval * wl + bl) : sinf(tval * S.wps[d - 1] + S.bps[d - 1]);
    }
    __syncthreads();

    // obs_emb + xhT = [x(33) | h(32) | pad]^T
    for (int i = tid; i < VV * DD; i += 256) {
      const int v = i >> 5, d = i & 31;
      const float ve = fmaxf(S.dats[v] * S.wvs[d] + S.bvs[d], 0.f);
      const float o  = (ve + S.tes[d] + S.emb1s[v][d]) * S.msk[v];
      S.obs[v][d]      = o;
      S.xhT[d][v]      = (__bf16)o;
      S.xhT[33 + d][v] = (__bf16)S.h[v][d];
    }
    if (tid < VV) S.xhT[32][tid] = (__bf16)S.rar[tid];
    // cur_adj = adj*(1 - rarity_W*|dr|)*m_i*m_j off-diag, 1 on diag
    for (int i = tid; i < VV * VV; i += 256) {
      const int r = i >> 6, c = i & 63;
      float val;
      if (r == c) val = 1.0f;
      else val = S.adj[r][c] * (1.0f - S.rarw[r][c] * fabsf(S.rar[r] - S.rar[c]))
                 * S.msk[r] * S.msk[c];
      S.adjb[r][c] = (__bf16)val;
    }
    __syncthreads();

    // comb = cur_adj(64x64) @ xh(64x80)  -> 4x5 tiles, K=64
    for (int tile = wave; tile < 20; tile += 8) {
      const int mo = (tile & 3) * 16;
      const int no = (tile >> 2) * 16;
      v8f acc = {};
      acc = wmma_bf16(ldsA(&S.adjb[0][0], VV, mo, 0),  ldsBT(&S.xhT[0][0], VV, 0,  no), acc);
      acc = wmma_bf16(ldsA(&S.adjb[0][0], VV, mo, 32), ldsBT(&S.xhT[0][0], VV, 32, no), acc);
      stC(&S.comb[0][0], 80, mo, no, acc);
    }
    __syncthreads();

    // combq[v][q*65+kk] = vv[v][q] * comb[v][kk]
    for (int i = tid; i < VV * 65; i += 256) {
      const int v = i / 65, kk = i - v * 65;
      const float c = S.comb[v][kk];
#pragma unroll
      for (int q = 0; q < QQ; ++q)
        S.combq[v][q * 65 + kk] = (__bf16)(S.vv[v][q] * c);
    }
    __syncthreads();

    // r,u gates: 2 x (4x2 tiles), K=352
    for (int tile = wave; tile < 16; tile += 8) {
      const int g  = tile >> 3;
      const int mo = (tile & 3) * 16;
      const int no = ((tile >> 2) & 1) * 16;
      const __bf16* Wg = g ? &S.wuT[0][0] : &S.wrT[0][0];
      const float* be  = g ? &S.beffu[0][0] : &S.beffr[0][0];
      float* dst       = g ? &S.gu[0][0] : &S.gr[0][0];
      v8f acc = {};
#pragma unroll
      for (int kc = 0; kc < KC; kc += 32)
        acc = wmma_bf16(ldsA(&S.combq[0][0], KC, mo, kc), ldsBT(Wg, KC, kc, no), acc);
      const int n  = no + (lane & 15);
      const int mb = mo + ((lane & 16) ? 8 : 0);
#pragma unroll
      for (int r = 0; r < 8; ++r) {
        const float x = acc[r] + be[(mb + r) * DD + n];
        dst[(mb + r) * DD + n] = 1.0f / (1.0f + expf(-x));
      }
    }
    __syncthreads();

    // combq2 from [x | r*h], scaled by vv (reuse combq)
    for (int i = tid; i < VV * 65; i += 256) {
      const int v = i / 65, kk = i - v * 65;
      float xv;
      if (kk < DD)       xv = S.obs[v][kk];
      else if (kk == DD) xv = S.rar[v];
      else               xv = S.gr[v][kk - 33] * S.h[v][kk - 33];
#pragma unroll
      for (int q = 0; q < QQ; ++q)
        S.combq[v][q * 65 + kk] = (__bf16)(S.vv[v][q] * xv);
    }
    __syncthreads();

    // cand gate: 4x2 tiles, K=352
    for (int tile = wave; tile < 8; tile += 8) {
      const int mo = (tile & 3) * 16;
      const int no = ((tile >> 2) & 1) * 16;
      float* gcf = &S.gc[0][0];
      v8f acc = {};
#pragma unroll
      for (int kc = 0; kc < KC; kc += 32)
        acc = wmma_bf16(ldsA(&S.combq[0][0], KC, mo, kc), ldsBT(&S.wcT[0][0], KC, kc, no), acc);
      const int n  = no + (lane & 15);
      const int mb = mo + ((lane & 16) ? 8 : 0);
#pragma unroll
      for (int r = 0; r < 8; ++r)
        gcf[(mb + r) * DD + n] = tanhf(acc[r] + S.beffc[mb + r][n]);
    }
    __syncthreads();

    // GRU update + snapshot
    const bool snap = (t == len - 1);
    for (int i = tid; i < VV * DD; i += 256) {
      const int v = i >> 5, d = i & 31;
      const float hr = S.gr[v][d] * S.h[v][d];
      const float u  = S.gu[v][d];
      const float hn = (1.0f - u) * hr + u * S.gc[v][d];
      const float hv = (S.msk[v] > 0.5f) ? hn : S.h[v][d];
      S.h[v][d] = hv;
      if (snap) S.hout[v][d] = hv;
    }
    __syncthreads();
  }

  if (tid < VV) {
    float s = 0.f;
#pragma unroll
    for (int d = 0; d < DD; ++d) s += S.hout[tid][d];
    g_outsum[b * VV + tid] = s;
  }
}

// ===========================================================================
// Kernel 3: classifier head.  feat(128x128) @ Wc1(128x200) on WMMA.
// ===========================================================================
struct __align__(32) FinSmem {
  __bf16 feat[BB][BB];       // 32 KB  (A, row-major)
  __bf16 w1T[HIDP][BB];      // 53 KB  (B transposed: [n][k])
  float  hbuf[BB][HIDP];     // 106 KB
};

__global__ __launch_bounds__(256, 1)
void k_fin(const float* g_outsum, const float* g_se,
           const float* Wc1, const float* bc1,
           const float* Wc2, const float* bc2, float* out) {
  extern __shared__ char smem_raw[];
  FinSmem& S = *(FinSmem*)smem_raw;
  const int tid  = threadIdx.x;
  const int wave = tid >> 5;
  const int lane = tid & 31;

  for (int i = tid; i < BB * BB; i += 256) {
    const int b = i >> 7, j = i & 127;
    const float v = (j < VV) ? g_outsum[b * VV + j] : g_se[b * VV + (j - VV)];
    ((__bf16*)S.feat)[i] = (__bf16)v;
  }
  for (int i = tid; i < HIDP * BB; i += 256) {
    const int n = i >> 7, k = i & 127;
    (&S.w1T[0][0])[i] = (__bf16)((n < HID) ? Wc1[k * HID + n] : 0.0f);
  }
  __syncthreads();

  for (int tile = wave; tile < 8 * 13; tile += 8) {   // M=128 (8), N=208 (13)
    const int mo = (tile & 7) * 16;
    const int no = (tile >> 3) * 16;
    v8f acc = {};
#pragma unroll
    for (int kc = 0; kc < BB; kc += 32)
      acc = wmma_bf16(ldsA(&S.feat[0][0], BB, mo, kc), ldsBT(&S.w1T[0][0], BB, kc, no), acc);
    const int n  = no + (lane & 15);
    const int mb = mo + ((lane & 16) ? 8 : 0);
#pragma unroll
    for (int r = 0; r < 8; ++r)
      S.hbuf[mb + r][n] = (n < HID) ? fmaxf(acc[r] + bc1[n], 0.0f) : 0.0f;
  }
  __syncthreads();

  // logits = hidden @ Wc2 + bc2  (128x2)
  {
    const int b = tid >> 1, c = tid & 1;   // 256 threads == 128*2 outputs
    float s = bc2[c];
    for (int k = 0; k < HID; ++k) s += S.hbuf[b][k] * Wc2[k * NCLS + c];
    out[b * NCLS + c] = s;
  }
}

// ===========================================================================
extern "C" void kernel_launch(void* const* d_in, const int* in_sizes, int n_in,
                              void* d_out, int out_size, void* d_ws, size_t ws_size,
                              hipStream_t stream) {
  (void)in_sizes; (void)n_in; (void)out_size; (void)ws_size;
  const float* P    = (const float*)d_in[0];
  const float* Pst  = (const float*)d_in[1];
  const float* Pavg = (const float*)d_in[2];
  const int*   Plen = (const int*)  d_in[3];
  const float* Ptim = (const float*)d_in[4];
  const float* plm  = (const float*)d_in[5];
  const float* w_val= (const float*)d_in[6];
  const float* b_val= (const float*)d_in[7];
  const float* w_per= (const float*)d_in[8];
  const float* b_per= (const float*)d_in[9];
  const float* w_lin= (const float*)d_in[10];
  const float* b_lin= (const float*)d_in[11];
  const float* emb1 = (const float*)d_in[12];
  const float* Ws1  = (const float*)d_in[13];
  const float* bs1  = (const float*)d_in[14];
  const float* Ws2  = (const float*)d_in[15];
  const float* bs2  = (const float*)d_in[16];
  const float* Wg1  = (const float*)d_in[17];
  const float* bg1  = (const float*)d_in[18];
  const float* Wg2  = (const float*)d_in[19];
  const float* bg2  = (const float*)d_in[20];
  const float* rarW = (const float*)d_in[21];
  const float* Wu   = (const float*)d_in[22];
  const float* bu   = (const float*)d_in[23];
  const float* Wr   = (const float*)d_in[24];
  const float* br   = (const float*)d_in[25];
  const float* Wc   = (const float*)d_in[26];
  const float* bc   = (const float*)d_in[27];
  const float* W_se = (const float*)d_in[28];
  const float* b_se = (const float*)d_in[29];
  const float* Wc1  = (const float*)d_in[30];
  const float* bc1  = (const float*)d_in[31];
  const float* Wc2  = (const float*)d_in[32];
  const float* bc2  = (const float*)d_in[33];

  // workspace carve (byte offsets, 2KB-aligned blocks)
  char* ws = (char*)d_ws;
  float*  g_vv     = (float*) (ws + 0);        // 64*5
  float*  g_adj    = (float*) (ws + 2048);     // 64*64
  float*  g_beffr  = (float*) (ws + 18432);    // 64*32
  float*  g_beffu  = (float*) (ws + 26624);
  float*  g_beffc  = (float*) (ws + 34816);
  float*  g_se     = (float*) (ws + 43008);    // 128*64
  float*  g_outsum = (float*) (ws + 75776);    // 128*64
  __bf16* g_wr     = (__bf16*)(ws + 108544);   // 32*352 (transposed)
  __bf16* g_wu     = (__bf16*)(ws + 131072);
  __bf16* g_wc     = (__bf16*)(ws + 153600);

  (void)hipFuncSetAttribute((const void*)k_pre,  hipFuncAttributeMaxDynamicSharedMemorySize, (int)sizeof(PreSmem));
  (void)hipFuncSetAttribute((const void*)k_main, hipFuncAttributeMaxDynamicSharedMemorySize, (int)sizeof(MainSmem));
  (void)hipFuncSetAttribute((const void*)k_fin,  hipFuncAttributeMaxDynamicSharedMemorySize, (int)sizeof(FinSmem));

  k_pre<<<1, 256, sizeof(PreSmem), stream>>>(
      plm, Ws1, bs1, Ws2, bs2, Wg1, bg1, Wg2, bg2,
      Wr, Wu, Wc, br, bu, bc, Pst, W_se, b_se,
      g_vv, g_adj, g_wr, g_wu, g_wc, g_beffr, g_beffu, g_beffc, g_se);

  k_main<<<BB, 256, sizeof(MainSmem), stream>>>(
      P, Pavg, Plen, Ptim, w_val, b_val, w_per, b_per, w_lin, b_lin,
      emb1, rarW, g_adj, g_vv, g_wr, g_wu, g_wc,
      g_beffr, g_beffu, g_beffc, g_outsum);

  k_fin<<<1, 256, sizeof(FinSmem), stream>>>(
      g_outsum, g_se, Wc1, bc1, Wc2, bc2, (float*)d_out);
}
